// PredictMiddleViewResidualGCN_60138132079165
// MI455X (gfx1250) — matrix operand order repaired and checked by reading, not compiled, not run on hardware
//
#include <hip/hip_runtime.h>
#include <hip/hip_bf16.h>
#include <stdint.h>

#define HID 128

typedef __attribute__((ext_vector_type(16))) __bf16 v16bf;
typedef __attribute__((ext_vector_type(8)))  __bf16 v8bf;
typedef __attribute__((ext_vector_type(8)))  float  v8f;

__device__ __forceinline__ __bf16 f2bf(float f) {
  // round-to-nearest-even f32 -> bf16 via integer ops (no cvt dependency)
  unsigned u = __builtin_bit_cast(unsigned, f);
  unsigned r = u + 0x7FFFu + ((u >> 16) & 1u);
  unsigned short hs = (unsigned short)(r >> 16);
  return __builtin_bit_cast(__bf16, hs);
}

// ---------------- degree / normalization ----------------
__global__ void k_deg_init(float* __restrict__ deg, int N) {
  int i = blockIdx.x * blockDim.x + threadIdx.x;
  if (i < N) deg[i] = 1.0f;  // self loop
}

__global__ void k_deg_count(const int* __restrict__ dst, float* __restrict__ deg, int E) {
  int e = blockIdx.x * blockDim.x + threadIdx.x;
  if (e < E) unsafeAtomicAdd(&deg[dst[e]], 1.0f);
}

__global__ void k_dis(const float* __restrict__ deg, float* __restrict__ dis, int N) {
  int i = blockIdx.x * blockDim.x + threadIdx.x;
  if (i < N) dis[i] = rsqrtf(deg[i]);  // deg >= 1 always
}

// ---------------- layer 0 dense: T = mask(x) @ W0  ([N,4]@[4,128]) ----------------
__global__ void k_layer0_dense(const float* __restrict__ x, const unsigned char* __restrict__ mask,
                               const float* __restrict__ W0, float* __restrict__ T, int N) {
  unsigned idx = blockIdx.x * blockDim.x + threadIdx.x;
  int n = idx >> 5;
  if (n >= N) return;
  int c4 = (idx & 31) * 4;
  float m = mask[n] ? 1.0f : 0.0f;
  float h[4];
#pragma unroll
  for (int k = 0; k < 4; ++k) h[k] = x[(size_t)n * 4 + k] * m;
#pragma unroll
  for (int j = 0; j < 4; ++j) {
    int c = c4 + j;
    float s = 0.0f;
#pragma unroll
    for (int k = 0; k < 4; ++k) s += h[k] * W0[k * HID + c];
    T[(size_t)n * HID + c] = s;
  }
}

// ---------------- AGG init with self-loop contribution: AGG = T * dis^2 ----------------
__global__ void k_selfloop(const float* __restrict__ T, const float* __restrict__ dis,
                           float* __restrict__ AGG, int N) {
  unsigned idx = blockIdx.x * blockDim.x + threadIdx.x;
  int n = idx >> 5;
  if (n >= N) return;
  int c4 = (idx & 31) * 4;
  float w = dis[n] * dis[n];
  size_t base = (size_t)n * HID + c4;
  float4 v = *(const float4*)(T + base);
  float4 o = make_float4(v.x * w, v.y * w, v.z * w, v.w * w);
  *(float4*)(AGG + base) = o;
}

// ---------------- edge scatter: AGG[dst] += T[src] * dis[src]*dis[dst] ----------------
// one wave = one edge, each lane a float4 chunk of the 128-wide feature row
__global__ void k_scatter(const int* __restrict__ src, const int* __restrict__ dst,
                          const float* __restrict__ dis, const float* __restrict__ T,
                          float* __restrict__ AGG, int E) {
  unsigned idx = blockIdx.x * blockDim.x + threadIdx.x;
  int e = idx >> 5;
  if (e >= E) return;
  int c4 = (idx & 31) * 4;
  int s = src[e], d = dst[e];
  float nrm = dis[s] * dis[d];
  float4 v = *(const float4*)(T + (size_t)s * HID + c4);
  float* out = AGG + (size_t)d * HID + c4;
  unsafeAtomicAdd(out + 0, v.x * nrm);
  unsafeAtomicAdd(out + 1, v.y * nrm);
  unsafeAtomicAdd(out + 2, v.z * nrm);
  unsafeAtomicAdd(out + 3, v.w * nrm);
}

// ---------------- layer 0 finish: H = relu(AGG + b0 + (h0@res_W + res_b)) ----------------
__global__ void k_finish0(const float* __restrict__ AGG, const float* __restrict__ b0,
                          const float* __restrict__ x, const unsigned char* __restrict__ mask,
                          const float* __restrict__ resW, const float* __restrict__ resb,
                          float* __restrict__ H, __bf16* __restrict__ Hbf, int N) {
  unsigned idx = blockIdx.x * blockDim.x + threadIdx.x;
  int n = idx >> 5;
  if (n >= N) return;
  int c4 = (idx & 31) * 4;
  float m = mask[n] ? 1.0f : 0.0f;
  float h0[4];
#pragma unroll
  for (int k = 0; k < 4; ++k) h0[k] = x[(size_t)n * 4 + k] * m;
  size_t base = (size_t)n * HID + c4;
#pragma unroll
  for (int j = 0; j < 4; ++j) {
    int c = c4 + j;
    float r = resb[c];
#pragma unroll
    for (int k = 0; k < 4; ++k) r += h0[k] * resW[k * HID + c];
    float v = fmaxf(AGG[base + j] + b0[c] + r, 0.0f);
    H[base + j] = v;
    Hbf[base + j] = f2bf(v);
  }
}

// ---------------- identity-residual finish: H = relu(AGG + b + H) ----------------
__global__ void k_finish(const float* __restrict__ AGG, const float* __restrict__ b,
                         float* __restrict__ H, __bf16* __restrict__ Hbf, int N) {
  unsigned idx = blockIdx.x * blockDim.x + threadIdx.x;
  int n = idx >> 5;
  if (n >= N) return;
  int c4 = (idx & 31) * 4;
  size_t base = (size_t)n * HID + c4;
  float4 a = *(const float4*)(AGG + base);
  float4 hv = *(const float4*)(H + base);
  float o0 = fmaxf(a.x + b[c4 + 0] + hv.x, 0.0f);
  float o1 = fmaxf(a.y + b[c4 + 1] + hv.y, 0.0f);
  float o2 = fmaxf(a.z + b[c4 + 2] + hv.z, 0.0f);
  float o3 = fmaxf(a.w + b[c4 + 3] + hv.w, 0.0f);
  *(float4*)(H + base) = make_float4(o0, o1, o2, o3);
  __bf16* hb = Hbf + base;
  hb[0] = f2bf(o0); hb[1] = f2bf(o1); hb[2] = f2bf(o2); hb[3] = f2bf(o3);
}

// ---------------- pack W [128,128] f32 into WMMA B-fragment bf16 layout ----------------
// Wpk[kt(4)][ct(8)][lane(32)][16]: lane -> col = ct*16 + lane%16,
// element j -> K = kt*32 + (lane>=16 ? 16 : 0) + j  (contiguous K order, same as A in k_gemm_wmma)
__global__ void k_packW(const float* __restrict__ W, __bf16* __restrict__ Wpk) {
  int t = blockIdx.x * blockDim.x + threadIdx.x;
  if (t >= 1024) return;
  int lane = t & 31;
  int ct = (t >> 5) & 7;
  int kt = t >> 8;
  int col = ct * 16 + (lane & 15);
  int kb = kt * 32 + ((lane >> 4) ? 16 : 0);
#pragma unroll
  for (int j = 0; j < 16; ++j)
    Wpk[(size_t)t * 16 + j] = f2bf(W[(size_t)(kb + j) * HID + col]);
}

// ---------------- WMMA GEMM: T[N,128] = Hbf[N,128] @ Wpk (f32 accumulate) ----------------
// block = 8 waves; wave w -> 16x16 C tile (rows blockIdx*16, cols w*16); K loop of 4 x 32
__global__ void k_gemm_wmma(const __bf16* __restrict__ Hbf, const __bf16* __restrict__ Wpk,
                            float* __restrict__ T, int N) {
  int wave = threadIdx.x >> 5;
  int lane = threadIdx.x & 31;
  int row0 = blockIdx.x * 16;
  int m = lane & 15;
  int kh = (lane >> 4) << 4;  // 0 or 16: contiguous K order matching Wpk
  const __bf16* hrow = Hbf + (size_t)(row0 + m) * HID;
  v8f acc = {};
#pragma unroll
  for (int kt = 0; kt < 4; ++kt) {
    int k0 = kt * 32;
    v8bf alo = *(const v8bf*)(hrow + k0 + kh);        // K = k0+kh .. +7
    v8bf ahi = *(const v8bf*)(hrow + k0 + kh + 8);    // K = k0+kh+8 .. +15
    v16bf a;
#pragma unroll
    for (int j = 0; j < 8; ++j) { a[j] = alo[j]; a[j + 8] = ahi[j]; }
    v16bf bfrag = *(const v16bf*)(Wpk + (((size_t)kt * 8 + wave) * 32 + lane) * 16);
    acc = __builtin_amdgcn_wmma_f32_16x16x32_bf16(false, a, false, bfrag,
                                                  (short)0, acc, false, false);
  }
  // C/D layout: VGPR i -> M = i + (lane>=16 ? 8 : 0), N = lane%16
  int mrow = (lane >> 4) * 8;
  float* out = T + (size_t)row0 * HID + wave * 16 + (lane & 15);
#pragma unroll
  for (int i = 0; i < 8; ++i)
    out[(size_t)(mrow + i) * HID] = acc[i];
}

// scalar tail (only used if N % 16 != 0; N=100000 divides evenly)
__global__ void k_gemm_tail(const float* __restrict__ H, const float* __restrict__ W,
                            float* __restrict__ T, int nStart, int N) {
  unsigned idx = blockIdx.x * blockDim.x + threadIdx.x;
  int r = nStart + (int)(idx / HID);
  int c = (int)(idx % HID);
  if (r >= N) return;
  float s = 0.0f;
  for (int k = 0; k < HID; ++k) s += H[(size_t)r * HID + k] * W[(size_t)k * HID + c];
  T[(size_t)r * HID + c] = s;
}

// ---------------- final FC: out = H @ fc_W + fc_b  ([N,128]@[128,4]) ----------------
__global__ void k_fc_out(const float* __restrict__ H, const float* __restrict__ fcW,
                         const float* __restrict__ fcb, float* __restrict__ out, int N) {
  unsigned idx = blockIdx.x * blockDim.x + threadIdx.x;
  if (idx >= (unsigned)N * 4u) return;
  int n = idx >> 2, oc = idx & 3;
  float s = fcb[oc];
  const float* hrow = H + (size_t)n * HID;
#pragma unroll 8
  for (int k = 0; k < HID; ++k) s += hrow[k] * fcW[k * 4 + oc];
  out[idx] = s;
}

__global__ void k_mask_out(const unsigned char* __restrict__ mask, float* __restrict__ out, int N) {
  int n = blockIdx.x * blockDim.x + threadIdx.x;
  if (n < N) out[(size_t)N * 4 + n] = mask[n] ? 1.0f : 0.0f;
}

// ---------------- host ----------------
extern "C" void kernel_launch(void* const* d_in, const int* in_sizes, int n_in,
                              void* d_out, int out_size, void* d_ws, size_t ws_size,
                              hipStream_t stream) {
  const float* x            = (const float*)d_in[0];
  const int* ei             = (const int*)d_in[1];
  const unsigned char* mask = (const unsigned char*)d_in[2];  // jax bool = 1 byte
  const float* W0   = (const float*)d_in[3];
  const float* b0   = (const float*)d_in[4];
  const float* resW = (const float*)d_in[5];
  const float* resb = (const float*)d_in[6];
  const float* W1   = (const float*)d_in[7];
  const float* b1   = (const float*)d_in[8];
  const float* W2   = (const float*)d_in[9];
  const float* b2   = (const float*)d_in[10];
  const float* fcW  = (const float*)d_in[11];
  const float* fcb  = (const float*)d_in[12];

  int N = in_sizes[0] / 4;
  int E = in_sizes[1] / 2;
  const int* src = ei;       // edge_index[0]
  const int* dst = ei + E;   // edge_index[1]

  char* wsb = (char*)d_ws;
  size_t off = 0;
  auto carve = [&](size_t bytes) -> char* {
    char* p = wsb + off;
    off = (off + bytes + 255) & ~(size_t)255;
    return p;
  };
  float*  deg = (float*)carve((size_t)N * 4);
  float*  dis = (float*)carve((size_t)N * 4);
  float*  H   = (float*)carve((size_t)N * HID * 4);
  float*  T   = (float*)carve((size_t)N * HID * 4);
  float*  AGG = (float*)carve((size_t)N * HID * 4);
  __bf16* Hbf = (__bf16*)carve((size_t)N * HID * 2);
  __bf16* Wp1 = (__bf16*)carve((size_t)HID * HID * 2);
  __bf16* Wp2 = (__bf16*)carve((size_t)HID * HID * 2);
  (void)ws_size; (void)n_in; (void)out_size;

  int gN  = (N + 255) / 256;
  int gE  = (E + 255) / 256;
  int gNC = (int)(((long long)N * 32 + 255) / 256);
  int gEC = (int)(((long long)E * 32 + 255) / 256);

  // normalization
  k_deg_init<<<gN, 256, 0, stream>>>(deg, N);
  k_deg_count<<<gE, 256, 0, stream>>>(dst, deg, E);
  k_dis<<<gN, 256, 0, stream>>>(deg, dis, N);

  // pre-pack weights into WMMA B-fragment layout
  k_packW<<<4, 256, 0, stream>>>(W1, Wp1);
  k_packW<<<4, 256, 0, stream>>>(W2, Wp2);

  // ---- layer 0 ----
  k_layer0_dense<<<gNC, 256, 0, stream>>>(x, mask, W0, T, N);
  k_selfloop<<<gNC, 256, 0, stream>>>(T, dis, AGG, N);
  k_scatter<<<gEC, 256, 0, stream>>>(src, dst, dis, T, AGG, E);
  k_finish0<<<gNC, 256, 0, stream>>>(AGG, b0, x, mask, resW, resb, H, Hbf, N);

  // ---- layers 1 & 2 ----
  int n16 = N / 16;
  for (int layer = 0; layer < 2; ++layer) {
    const __bf16* Wp = (layer == 0) ? Wp1 : Wp2;
    const float*  Wf = (layer == 0) ? W1  : W2;
    const float*  bb = (layer == 0) ? b1  : b2;
    if (n16 > 0) k_gemm_wmma<<<n16, 256, 0, stream>>>(Hbf, Wp, T, N);
    int tail = N - n16 * 16;
    if (tail > 0)
      k_gemm_tail<<<(tail * HID + 255) / 256, 256, 0, stream>>>(H, Wf, T, n16 * 16, N);
    k_selfloop<<<gNC, 256, 0, stream>>>(T, dis, AGG, N);
    k_scatter<<<gEC, 256, 0, stream>>>(src, dst, dis, T, AGG, E);
    k_finish<<<gNC, 256, 0, stream>>>(AGG, bb, H, Hbf, N);
  }

  // ---- output: x_out then mask ----
  float* out = (float*)d_out;
  k_fc_out<<<((N * 4) + 255) / 256, 256, 0, stream>>>(H, fcW, fcb, out, N);
  k_mask_out<<<gN, 256, 0, stream>>>(mask, out, N);
}
